// CapsuleGNNModel_69544110457563
// MI455X (gfx1250) — matrix operand order repaired and checked by reading, not compiled, not run on hardware
//
#include <hip/hip_runtime.h>

#define NN 30000
#define FF 256
#define EE 480000
#define ET 510000          // E + N self loops
#define HEADS 4
#define HDIM 32
#define HH 128             // HEADS*HDIM
#define NB 118             // ceil(NN/256)

typedef __attribute__((ext_vector_type(16))) __bf16 v16bf;
typedef __attribute__((ext_vector_type(8)))  float  v8f;
typedef __attribute__((ext_vector_type(4)))  unsigned int u32x4;

// ---------- helpers ----------
__device__ __forceinline__ unsigned short f2bf(float f) {
  unsigned u = __float_as_uint(f);
  unsigned r = u + 0x7FFFu + ((u >> 16) & 1u);   // round-to-nearest-even
  return (unsigned short)(r >> 16);
}
__device__ __forceinline__ int edge_dst(const int* ei, int e) {
  return (e < EE) ? ei[EE + e] : (e - EE);
}
__device__ __forceinline__ int edge_src(const int* ei, int e) {
  return (e < EE) ? ei[e] : (e - EE);
}

// ---------- 1. BatchNorm ----------
__global__ void bn_stats(const float* __restrict__ x, float* __restrict__ csum,
                         float* __restrict__ csq) {
  int col = threadIdx.x;                 // 256 threads = F columns
  float s = 0.f, s2 = 0.f;
  for (int row = blockIdx.x; row < NN; row += gridDim.x) {
    float v = x[(size_t)row * FF + col];
    s += v; s2 += v * v;
  }
  atomicAdd(&csum[col], s);
  atomicAdd(&csq[col], s2);
}

__global__ void bn_finalize(const float* __restrict__ csum, const float* __restrict__ csq,
                            float* __restrict__ mean, float* __restrict__ invstd) {
  int c = threadIdx.x;
  float m = csum[c] * (1.f / NN);
  float var = csq[c] * (1.f / NN) - m * m;
  mean[c] = m;
  invstd[c] = rsqrtf(var + 1e-5f);
}

__global__ void bn_apply(const float* __restrict__ x, const float* __restrict__ mean,
                         const float* __restrict__ invstd, const float* __restrict__ gamma,
                         const float* __restrict__ beta, unsigned short* __restrict__ xb16) {
  size_t i = (size_t)blockIdx.x * blockDim.x + threadIdx.x;
  int c = (int)(i & (FF - 1));
  float v = (x[i] - mean[c]) * invstd[c] * gamma[c] + beta[c];
  xb16[i] = f2bf(v);
}

// ---------- 2. Pack weights into WMMA B-fragment layout ----------
// B 32x16 bf16, this code's contiguous k order (same as A): lanes0-15 hold K=0..15 (elem j = K j), lanes16-31 hold K=16..31.
// Storage: elem = ((kt*8+nt)*32+lane)*16 + j -> contiguous 32B per lane.
__global__ void pack_weights(const float* __restrict__ WL, const float* __restrict__ WG,
                             unsigned short* __restrict__ PL, unsigned short* __restrict__ PG) {
  int gid = blockIdx.x * blockDim.x + threadIdx.x;   // 65536 total
  int m = gid >> 15;
  int e = gid & 32767;
  int j = e & 15;
  int lane = (e >> 4) & 31;
  int nt = (e >> 9) & 7;
  int kt = e >> 12;
  int n = nt * 16 + (lane & 15);
  int k = kt * 32 + ((lane & 16) ? 16 : 0) + j;
  const float* W = m ? WG : WL;
  unsigned short* P = m ? PG : PL;
  P[e] = f2bf(W[k * HH + n]);
}

// ---------- 3. Dual GEMM: one wave = 16x64 stripe of both outputs (no spills) ----------
__global__ void __launch_bounds__(32, 1)
gemm_dual(const unsigned short* __restrict__ A16,
          const unsigned short* __restrict__ PL,
          const unsigned short* __restrict__ PG,
          const float* __restrict__ b_lin,
          float* __restrict__ hidden, float* __restrict__ xt) {
  int lane = threadIdx.x;
  int mt = blockIdx.x;
  int half = blockIdx.y;                 // N-half: nt in [half*4, half*4+4)
  int row = mt * 16 + (lane & 15);
  // A 16x32 bf16, this code's contiguous k order (same as B): lanes0-15 K{0..15}, lanes16-31 K{16..31}
  size_t abase = (size_t)row * FF + ((lane & 16) ? 16 : 0);
  v8f cl[4], cg[4];
#pragma unroll
  for (int nt = 0; nt < 4; ++nt) { cl[nt] = (v8f){}; cg[nt] = (v8f){}; }
#pragma unroll
  for (int kt = 0; kt < 8; ++kt) {
    union { u32x4 q[2]; v16bf v; } fa;
    const unsigned short* ap = A16 + abase + kt * 32;
    fa.q[0] = *(const u32x4*)(ap);
    fa.q[1] = *(const u32x4*)(ap + 8);
#pragma unroll
    for (int nt = 0; nt < 4; ++nt) {
      union { u32x4 q[2]; v16bf v; } fl, fg;
      int boff = ((kt * 8 + half * 4 + nt) * 32 + lane) * 16;
      fl.q[0] = *(const u32x4*)(PL + boff);
      fl.q[1] = *(const u32x4*)(PL + boff + 8);
      fg.q[0] = *(const u32x4*)(PG + boff);
      fg.q[1] = *(const u32x4*)(PG + boff + 8);
      cl[nt] = __builtin_amdgcn_wmma_f32_16x16x32_bf16(false, fa.v, false, fl.v,
                                                       (short)0, cl[nt], false, false);
      cg[nt] = __builtin_amdgcn_wmma_f32_16x16x32_bf16(false, fa.v, false, fg.v,
                                                       (short)0, cg[nt], false, false);
    }
  }
  // C layout: VGPR v -> M = v (+8 for lanes 16-31), N = lane&15
  int r0 = mt * 16 + ((lane & 16) ? 8 : 0);
#pragma unroll
  for (int nt = 0; nt < 4; ++nt) {
    int col = (half * 4 + nt) * 16 + (lane & 15);
    float bias = b_lin[col];
#pragma unroll
    for (int v = 0; v < 8; ++v) {
      hidden[(size_t)(r0 + v) * HH + col] = cl[nt][v] + bias;
      xt[(size_t)(r0 + v) * HH + col] = cg[nt][v];
    }
  }
}

// ---------- 4. GAT: per-edge alpha (leaky relu), then CSR gather ----------
__global__ void gat_alpha(const float* __restrict__ xt, const float* __restrict__ att,
                          const int* __restrict__ ei, float* __restrict__ alpha) {
  int gid = blockIdx.x * blockDim.x + threadIdx.x;
  if (gid >= ET * HEADS) return;
  int e = gid >> 2, h = gid & 3;
  int s = edge_src(ei, e), d = edge_dst(ei, e);
  const float* xi = xt + (size_t)d * HH + h * HDIM;
  const float* xj = xt + (size_t)s * HH + h * HDIM;
  const float* ai = att + h * (2 * HDIM);
  const float* aj = ai + HDIM;
  float acc = 0.f;
#pragma unroll
  for (int k = 0; k < HDIM; ++k) acc += xi[k] * ai[k] + xj[k] * aj[k];
  alpha[gid] = acc >= 0.f ? acc : 0.2f * acc;    // leaky relu
}

// CSR build: histogram -> 2-level exclusive scan -> fill
__global__ void deg_hist(const int* __restrict__ ei, int* __restrict__ deg) {
  int e = blockIdx.x * blockDim.x + threadIdx.x;
  if (e >= ET) return;
  atomicAdd(&deg[edge_dst(ei, e)], 1);
}

__global__ void scan_local(const int* __restrict__ deg, int* __restrict__ part,
                           int* __restrict__ bsum) {
  __shared__ int s[256];
  int tid = threadIdx.x;
  int n = blockIdx.x * 256 + tid;
  int v = (n < NN) ? deg[n] : 0;
  s[tid] = v;
  __syncthreads();
  for (int ofs = 1; ofs < 256; ofs <<= 1) {
    int t = (tid >= ofs) ? s[tid - ofs] : 0;
    __syncthreads();
    s[tid] += t;
    __syncthreads();
  }
  if (n < NN) part[n] = s[tid] - v;              // exclusive
  if (tid == 255) bsum[blockIdx.x] = s[255];
}

__global__ void scan_bsum(const int* __restrict__ bsum, int* __restrict__ bsex) {
  __shared__ int s[128];
  int tid = threadIdx.x;
  int v = (tid < NB) ? bsum[tid] : 0;
  s[tid] = v;
  __syncthreads();
  for (int ofs = 1; ofs < 128; ofs <<= 1) {
    int t = (tid >= ofs) ? s[tid - ofs] : 0;
    __syncthreads();
    s[tid] += t;
    __syncthreads();
  }
  bsex[tid] = s[tid] - v;                        // exclusive
}

__global__ void scan_add(const int* __restrict__ part, const int* __restrict__ bsex,
                         int* __restrict__ rowstart) {
  int n = blockIdx.x * blockDim.x + threadIdx.x;
  if (n < NN) rowstart[n] = part[n] + bsex[n >> 8];
  if (n == 0) rowstart[NN] = ET;
}

__global__ void csr_fill(const int* __restrict__ ei, const int* __restrict__ rowstart,
                         int* __restrict__ fill, int* __restrict__ eid) {
  int e = blockIdx.x * blockDim.x + threadIdx.x;
  if (e >= ET) return;
  int d = edge_dst(ei, e);
  int p = atomicAdd(&fill[d], 1);
  eid[rowstart[d] + p] = e;
}

// Gather aggregation: one wave per (node, head); lane = feature dim. No atomics.
__global__ void gat_gather(const int* __restrict__ ei, const int* __restrict__ rowstart,
                           const int* __restrict__ eid, const float* __restrict__ alpha,
                           const float* __restrict__ xt, float* __restrict__ aggr) {
  int n = blockIdx.x;
  int h = threadIdx.x >> 5;
  int lane = threadIdx.x & 31;
  int beg = rowstart[n], end = rowstart[n + 1];
  // segment max over this node's incoming edges
  float m = -1e30f;
  for (int i = beg + lane; i < end; i += 32) m = fmaxf(m, alpha[eid[i] * HEADS + h]);
  m = fmaxf(m, __shfl_xor(m, 1));  m = fmaxf(m, __shfl_xor(m, 2));
  m = fmaxf(m, __shfl_xor(m, 4));  m = fmaxf(m, __shfl_xor(m, 8));
  m = fmaxf(m, __shfl_xor(m, 16));
  // exp-sum + weighted message sum (coalesced xt row loads)
  float se = 0.f, acc = 0.f;
  for (int i = beg; i < end; ++i) {
    int e = eid[i];
    float w = __expf(alpha[e * HEADS + h] - m);
    se += w;
    int s = edge_src(ei, e);
    acc += w * xt[(size_t)s * HH + h * HDIM + lane];
  }
  aggr[(size_t)n * HH + h * HDIM + lane] = acc / (se + 1e-16f);
}

// ---------- 5. Capsule + routing: one wave per node, lane = o ----------
#define WAVES 8
__global__ void capsule_kernel(const float* __restrict__ aggr, const float* __restrict__ hidden,
                               const float* __restrict__ b_gat, const float* __restrict__ Wp,
                               const float* __restrict__ bp, const float* __restrict__ route_w,
                               float* __restrict__ out_norm, float* __restrict__ out_feat) {
  __shared__ float xcS[WAVES][8][32];
  __shared__ float uS[WAVES][16][8];
  __shared__ float lgS[WAVES][6][16];
  __shared__ float onS[WAVES][6];
  int lane = threadIdx.x & 31;
  int w = threadIdx.x >> 5;
  int n = blockIdx.x * WAVES + w;          // 30000 = 3750 * 8 exact

#pragma unroll
  for (int r = 0; r < 8; ++r) {
    float v = (r < 4) ? (aggr[(size_t)n * HH + r * 32 + lane] + b_gat[r * 32 + lane])
                      : hidden[(size_t)n * HH + (r - 4) * 32 + lane];
    xcS[w][r][lane] = v > 0.f ? v : 0.f;
  }
  __syncthreads();

#pragma unroll
  for (int t = 0; t < 4; ++t) {
    int idx = t * 32 + lane;
    int c = idx >> 6, r = (idx >> 3) & 7, o = idx & 7;
    float acc = bp[c * 8 + o];
    for (int h = 0; h < 32; ++h) acc += xcS[w][r][h] * Wp[(c * 32 + h) * 8 + o];
    float s = acc * acc;
    s += __shfl_xor(s, 1); s += __shfl_xor(s, 2); s += __shfl_xor(s, 4);
    uS[w][idx >> 3][o] = acc * (s / (1.f + s)) * rsqrtf(s);
  }
  for (int j = lane; j < 96; j += 32) lgS[w][j >> 4][j & 15] = 0.f;
  __syncthreads();

  float pr[6][16];
#pragma unroll
  for (int cc = 0; cc < 6; ++cc)
#pragma unroll
    for (int r = 0; r < 16; ++r) {
      float acc = 0.f;
#pragma unroll
      for (int i = 0; i < 8; ++i)
        acc += uS[w][r][i] * route_w[(((cc * 16 + r) * 8) + i) * 32 + lane];
      pr[cc][r] = acc;
    }

  float v6[6];
  for (int it = 0; it < 3; ++it) {
#pragma unroll
    for (int cc = 0; cc < 6; ++cc) {
      float mx = -1e30f;
      for (int r = 0; r < 16; ++r) mx = fmaxf(mx, lgS[w][cc][r]);
      float p[16], se = 0.f;
      for (int r = 0; r < 16; ++r) { p[r] = __expf(lgS[w][cc][r] - mx); se += p[r]; }
      float inv = 1.f / se;
      float acc = 0.f;
      for (int r = 0; r < 16; ++r) acc += p[r] * inv * pr[cc][r];
      float s = acc * acc;
      s += __shfl_xor(s, 1); s += __shfl_xor(s, 2); s += __shfl_xor(s, 4);
      s += __shfl_xor(s, 8); s += __shfl_xor(s, 16);
      float sn = s / (1.f + s);
      v6[cc] = acc * sn * rsqrtf(s);
      if (lane == 0) onS[w][cc] = sn;            // ||squash(s)|| = s/(1+s)
    }
    if (it < 2) {
#pragma unroll
      for (int cc = 0; cc < 6; ++cc)
        for (int r = 0; r < 16; ++r) {
          float d = pr[cc][r] * v6[cc];
          d += __shfl_xor(d, 1); d += __shfl_xor(d, 2); d += __shfl_xor(d, 4);
          d += __shfl_xor(d, 8); d += __shfl_xor(d, 16);
          if (lane == 0) lgS[w][cc][r] += d;
        }
    }
  }
#pragma unroll
  for (int cc = 0; cc < 6; ++cc)
    out_feat[(size_t)n * 192 + cc * 32 + lane] = v6[cc];
  if (lane < 6) out_norm[(size_t)n * 6 + lane] = onS[w][lane];
}

// ---------- 6. Loss ----------
__global__ void loss_kernel(const float* __restrict__ outp, const int* __restrict__ y,
                            float* __restrict__ loss) {
  int n = blockIdx.x * blockDim.x + threadIdx.x;
  if (n >= NN) return;
  float mx = -1e30f;
  for (int c = 0; c < 6; ++c) mx = fmaxf(mx, outp[n * 6 + c]);
  float se = 0.f;
  for (int c = 0; c < 6; ++c) se += __expf(outp[n * 6 + c] - mx);
  float lse = mx + logf(se);
  float lp = outp[n * 6 + y[n]] - lse;
  atomicAdd(loss, -lp * (1.f / NN));
}

// ---------- launch ----------
extern "C" void kernel_launch(void* const* d_in, const int* in_sizes, int n_in,
                              void* d_out, int out_size, void* d_ws, size_t ws_size,
                              hipStream_t stream) {
  const float* x      = (const float*)d_in[0];
  const float* gamma  = (const float*)d_in[1];
  const float* beta   = (const float*)d_in[2];
  const float* W_lin  = (const float*)d_in[3];
  const float* b_lin  = (const float*)d_in[4];
  const float* W_gat  = (const float*)d_in[5];
  const float* att    = (const float*)d_in[6];
  const float* b_gat  = (const float*)d_in[7];
  const float* Wp     = (const float*)d_in[8];
  const float* bp     = (const float*)d_in[9];
  const float* route_w= (const float*)d_in[10];
  const int*   ei     = (const int*)d_in[11];
  const int*   y      = (const int*)d_in[12];

  float* out_norm = (float*)d_out;                 // 30000*6
  float* out_loss = out_norm + (size_t)NN * 6;     // 1
  float* out_feat = out_loss + 1;                  // 30000*192

  char* base = (char*)d_ws;
  size_t off = 0;
  auto alloc = [&](size_t b) -> char* {
    char* p = base + off; off = (off + b + 255) & ~(size_t)255; return p;
  };
  float* colsum   = (float*)alloc(FF * 4);
  float* colsumsq = (float*)alloc(FF * 4);
  float* meanp    = (float*)alloc(FF * 4);
  float* invstd   = (float*)alloc(FF * 4);
  unsigned short* xb16  = (unsigned short*)alloc((size_t)NN * FF * 2);
  unsigned short* packL = (unsigned short*)alloc((size_t)FF * HH * 2);
  unsigned short* packG = (unsigned short*)alloc((size_t)FF * HH * 2);
  float* hidden = (float*)alloc((size_t)NN * HH * 4);
  float* xt     = (float*)alloc((size_t)NN * HH * 4);
  float* alpha  = (float*)alloc((size_t)ET * HEADS * 4);
  float* aggr   = (float*)alloc((size_t)NN * HH * 4);
  int* deg      = (int*)alloc((size_t)NN * 4);
  int* part     = (int*)alloc((size_t)NN * 4);
  int* bsum     = (int*)alloc(128 * 4);
  int* bsex     = (int*)alloc(128 * 4);
  int* rowstart = (int*)alloc((size_t)(NN + 1) * 4);
  int* fill     = (int*)alloc((size_t)NN * 4);
  int* eid      = (int*)alloc((size_t)ET * 4);

  hipMemsetAsync(colsum, 0, FF * 4 * 2, stream);   // colsum+colsumsq contiguous
  hipMemsetAsync(deg,  0, (size_t)NN * 4, stream);
  hipMemsetAsync(fill, 0, (size_t)NN * 4, stream);
  hipMemsetAsync(out_loss, 0, 4, stream);

  // dense pipeline
  bn_stats<<<300, 256, 0, stream>>>(x, colsum, colsumsq);
  bn_finalize<<<1, 256, 0, stream>>>(colsum, colsumsq, meanp, invstd);
  bn_apply<<<(NN * FF) / 256, 256, 0, stream>>>(x, meanp, invstd, gamma, beta, xb16);
  pack_weights<<<(2 * FF * HH) / 256, 256, 0, stream>>>(W_lin, W_gat, packL, packG);
  gemm_dual<<<dim3(NN / 16, 2), 32, 0, stream>>>(xb16, packL, packG, b_lin, hidden, xt);

  // CSR build
  int eg = (ET + 255) / 256;
  deg_hist<<<eg, 256, 0, stream>>>(ei, deg);
  scan_local<<<NB, 256, 0, stream>>>(deg, part, bsum);
  scan_bsum<<<1, 128, 0, stream>>>(bsum, bsex);
  scan_add<<<NB, 256, 0, stream>>>(part, bsex, rowstart);
  csr_fill<<<eg, 256, 0, stream>>>(ei, rowstart, fill, eid);

  // GAT attention (gather, no float atomics)
  int ag = (ET * HEADS + 255) / 256;
  gat_alpha<<<ag, 256, 0, stream>>>(xt, att, ei, alpha);
  gat_gather<<<NN, HEADS * 32, 0, stream>>>(ei, rowstart, eid, alpha, xt, aggr);

  // capsule head + loss
  capsule_kernel<<<NN / WAVES, WAVES * 32, 0, stream>>>(aggr, hidden, b_gat, Wp, bp,
                                                        route_w, out_norm, out_feat);
  loss_kernel<<<(NN + 255) / 256, 256, 0, stream>>>(out_norm, y, out_loss);
}
